// QAttention4D_39410619908249
// MI455X (gfx1250) — compile-verified, matrix operand order checked by reading, not run on hardware
//
#include <hip/hip_runtime.h>
#include <hip/hip_bf16.h>

typedef __attribute__((ext_vector_type(8))) int v8i;

#define WMMA_IU8(a, b, c) \
  __builtin_amdgcn_wmma_i32_16x16x64_iu8(true, (a), true, (b), (c), false, false)

static constexpr int Bb = 8, Nn = 577, Cc = 768, Hh = 12, Dd = 64;
static constexpr int M_ROWS = Bb * Nn;   // 4616
static constexpr int OC3 = 3 * Cc;       // 2304
static constexpr float EPSF = 1.1920929e-07f;

// ---- workspace layout (byte offsets, 256-aligned) ----
static constexpr size_t OFF_SCAL = 0;                                   // 16 uint slots
static constexpr size_t OFF_X8   = 256;                                 // 4616*768 int8
static constexpr size_t OFF_WQ8  = OFF_X8  + (size_t)M_ROWS * Cc;       // 2304*768 int8
static constexpr size_t OFF_WP8  = OFF_WQ8 + (size_t)OC3 * Cc;          // 768*768 int8
static constexpr size_t OFF_SWQ  = OFF_WP8 + (size_t)Cc * Cc;           // 2304 f32
static constexpr size_t OFF_SWP  = OFF_SWQ + (size_t)OC3 * 4;           // 768 f32
static constexpr size_t OFF_QKVI = OFF_SWP + (size_t)Cc * 4;            // 4616*2304 i32
static constexpr size_t OFF_QKV8 = OFF_QKVI + (size_t)M_ROWS * OC3 * 4; // 4616*2304 int8
static constexpr size_t OFF_CTXI = OFF_QKV8 + (size_t)M_ROWS * OC3;     // 4616*768 i32
static constexpr size_t OFF_CTX8 = OFF_CTXI + (size_t)M_ROWS * Cc * 4;  // 4616*768 int8

// slot indices in the scalar area
// 0: absmax(x) float-bits   1: s_x
// 2: absmax(qkv_f) bits     3: s_qkv
// 4: maxabs(ctx_int) uint   5: s_ctx

// ===================== fragment helpers (IU8 16x16x64 layout) =====================
// A (16x64, 8-bit): lane m=lane&15 holds K chunks {hi*8, 16+hi*8, 32+hi*8, 48+hi*8}
// B (64x16, 8-bit): lane n=lane&15 = column; same K chunk pattern along the row of W.
union FragU { v8i v; unsigned long long u[4]; };

__device__ __forceinline__ v8i frag_load_g(const signed char* base, long ld,
                                           int row0, int k0, int rowMax) {
  int lane = threadIdx.x & 31;
  int r = row0 + (lane & 15);
  if (r > rowMax) r = rowMax;           // clamp (harmless duplicate work at edges)
  int hi = (lane >> 4) * 8;
  const signed char* p = base + (long)r * ld + k0 + hi;
  FragU f;
  f.u[0] = *(const unsigned long long*)(p);
  f.u[1] = *(const unsigned long long*)(p + 16);
  f.u[2] = *(const unsigned long long*)(p + 32);
  f.u[3] = *(const unsigned long long*)(p + 48);
  return f.v;
}

__device__ __forceinline__ v8i frag_load_lds(const signed char* base, int ld,
                                             int row0, int k0) {
  int lane = threadIdx.x & 31;
  int r = row0 + (lane & 15);
  int hi = (lane >> 4) * 8;
  const signed char* p = base + r * ld + k0 + hi;
  FragU f;
  f.u[0] = *(const unsigned long long*)(p);
  f.u[1] = *(const unsigned long long*)(p + 16);
  f.u[2] = *(const unsigned long long*)(p + 32);
  f.u[3] = *(const unsigned long long*)(p + 48);
  return f.v;
}

// ===================== small kernels =====================
__global__ void k_init(unsigned* scal) {
  if (threadIdx.x < 16) scal[threadIdx.x] = 0u;
}

__global__ void k_absmax(const float* __restrict__ p, long n, unsigned* slot) {
  long i = (long)blockIdx.x * blockDim.x + threadIdx.x;
  long stride = (long)gridDim.x * blockDim.x;
  float m = 0.f;
  for (; i < n; i += stride) m = fmaxf(m, fabsf(p[i]));
  for (int s = 16; s; s >>= 1) m = fmaxf(m, __shfl_xor(m, s, 32));
  if ((threadIdx.x & 31) == 0) atomicMax(slot, __float_as_uint(m));
}

// v = rawIsInt ? (float)*raw : bits(*raw); v *= (*prescale) * cmul; out = max(v/127, eps)
__global__ void k_make_scale(const unsigned* raw, int rawIsInt,
                             const float* prescale, float cmul, float* out) {
  float v = rawIsInt ? (float)(*raw) : __uint_as_float(*raw);
  if (prescale) v *= *prescale;
  v *= cmul;
  *out = fmaxf(v / 127.f, EPSF);
}

__global__ void k_quant_act(const float* __restrict__ x, signed char* __restrict__ x8,
                            const float* s, long n) {
  float sc = *s;
  long i = (long)blockIdx.x * blockDim.x + threadIdx.x;
  long stride = (long)gridDim.x * blockDim.x;
  for (; i < n; i += stride) {
    float q = rintf(x[i] / sc);                 // round half-to-even, matches jnp.round
    q = fminf(fmaxf(q, -128.f), 127.f);
    x8[i] = (signed char)q;
  }
}

__global__ void k_wquant(const float* __restrict__ w, signed char* __restrict__ w8,
                         float* __restrict__ s_w, int rows, int K) {
  int wv = threadIdx.x >> 5, lane = threadIdx.x & 31;
  int r = blockIdx.x * 8 + wv;
  if (r >= rows) return;
  const float* wr = w + (long)r * K;
  float m = 0.f;
  for (int c = lane; c < K; c += 32) m = fmaxf(m, fabsf(wr[c]));
  for (int s = 16; s; s >>= 1) m = fmaxf(m, __shfl_xor(m, s, 32));
  float sc = fmaxf(m / 127.f, EPSF);
  if (lane == 0) s_w[r] = sc;
  for (int c = lane; c < K; c += 32) {
    float q = rintf(wr[c] / sc);
    q = fminf(fmaxf(q, -128.f), 127.f);
    w8[(long)r * K + c] = (signed char)q;
  }
}

// fixedpoint_mul hard path: z_int already integer-valued; requantize onto z_s
__global__ void k_requant(const int* __restrict__ zi, signed char* __restrict__ out8,
                          const float* s_act, const float* s_w, float cmul,
                          const float* s_out, long total, int ncols) {
  float sa = *s_act, zs = *s_out;
  long i = (long)blockIdx.x * blockDim.x + threadIdx.x;
  long stride = (long)gridDim.x * blockDim.x;
  for (; i < total; i += stride) {
    int o = (int)(i % ncols);
    float pre = sa * (s_w ? s_w[o] : cmul);
    float ns = pre / zs;
    float e = floorf(log2f(ns)) + 1.f;                       // frexp exponent
    float mant = floorf(ns * exp2f(-e) * 2147483648.f + 0.5f);
    float v = rintf((float)zi[i] * mant * exp2f(e - 31.f));
    v = fminf(fmaxf(v, -128.f), 127.f);
    out8[i] = (signed char)v;
  }
}

// ===================== int8 WMMA GEMM =====================
// D[m,n] = sum_k A[m,k]*W[n,k] + round(bias[n]/(sa*sw[n]))
// mode 0: write int32 + fused max|v*acc_s| -> maxslot ; mode 1: write float v*acc_s
__global__ __launch_bounds__(256) void k_gemm_i8(
    const signed char* __restrict__ A, const signed char* __restrict__ W,
    int M, int N, int K,
    const float* __restrict__ bias, const float* __restrict__ s_w,
    const float* s_act, int* __restrict__ out_i, float* __restrict__ out_f,
    unsigned* maxslot, int mode) {
  int w = threadIdx.x >> 5, lane = threadIdx.x & 31;
  int wm = w >> 2, wn = w & 3;
  int m0 = blockIdx.x * 64 + wm * 32;
  int n0 = blockIdx.y * 128 + wn * 32;

  v8i acc00 = {}, acc01 = {}, acc10 = {}, acc11 = {};
  for (int k0 = 0; k0 < K; k0 += 64) {
    v8i a0 = frag_load_g(A, K, m0, k0, M - 1);
    v8i a1 = frag_load_g(A, K, m0 + 16, k0, M - 1);
    v8i b0 = frag_load_g(W, K, n0, k0, N - 1);
    v8i b1 = frag_load_g(W, K, n0 + 16, k0, N - 1);
    acc00 = WMMA_IU8(a0, b0, acc00);
    acc01 = WMMA_IU8(a0, b1, acc01);
    acc10 = WMMA_IU8(a1, b0, acc10);
    acc11 = WMMA_IU8(a1, b1, acc11);
  }

  float sa = *s_act;
  float localmax = 0.f;
  int rb = (lane >> 4) * 8;
#pragma unroll
  for (int ti = 0; ti < 2; ++ti) {
#pragma unroll
    for (int tj = 0; tj < 2; ++tj) {
      v8i acc = ti ? (tj ? acc11 : acc10) : (tj ? acc01 : acc00);
      int col = n0 + tj * 16 + (lane & 15);
      float accs = sa * s_w[col];
      int bi = (int)rintf(bias[col] / accs);
#pragma unroll
      for (int i = 0; i < 8; ++i) {
        int row = m0 + ti * 16 + rb + i;      // C/D layout: vgpr i -> M = i (+8 hi half)
        if (row < M) {
          int v = acc[i] + bi;
          if (mode == 0) {
            out_i[(long)row * N + col] = v;
            localmax = fmaxf(localmax, fabsf((float)v * accs));
          } else {
            out_f[(long)row * N + col] = (float)v * accs;
          }
        }
      }
    }
  }
  if (mode == 0) {
    for (int s = 16; s; s >>= 1) localmax = fmaxf(localmax, __shfl_xor(localmax, s, 32));
    if (lane == 0) atomicMax(maxslot, __float_as_uint(localmax));
  }
}

// ===================== fused integer attention =====================
// grid: (37 row tiles, H, B); block 256 (8 waves). Per block: 16 query rows.
__global__ __launch_bounds__(256) void k_attn(
    const signed char* __restrict__ qkv8, int* __restrict__ ctx_i,
    const float* s_qkv_slot, unsigned* maxslot) {
  __shared__ __align__(16) unsigned char smemA[64 * 640];    // logits(int) -> exp(f32) -> V^T(int8)
  __shared__ __align__(16) signed char  sProb[16 * 640];     // int8 softmax probs, K-padded

  int rt = blockIdx.x, h = blockIdx.y, b = blockIdx.z;
  int w = threadIdx.x >> 5, lane = threadIdx.x & 31;
  const long headQ = (long)b * Nn * OC3 + (long)h * Dd;
  const signed char* qbase = qkv8 + headQ;                   // i=0
  const signed char* kbase = qkv8 + headQ + Cc;              // i=1
  const signed char* vbase = qkv8 + headQ + 2 * Cc;          // i=2

  // ---- phase 1: logits = q . k^T  (one IU8 WMMA per 16-col tile, K=D=64) ----
  int* L = (int*)smemA;                                      // [16][608]
  {
    v8i aq = frag_load_g(qbase, OC3, rt * 16, 0, Nn - 1);
    int rb = (lane >> 4) * 8;
    for (int ct = w; ct < 37; ct += 8) {
      v8i bk = frag_load_g(kbase, OC3, ct * 16, 0, Nn - 1);
      v8i z = {};
      v8i d = WMMA_IU8(aq, bk, z);
      int cc = ct * 16 + (lane & 15);
#pragma unroll
      for (int i = 0; i < 8; ++i) L[(rb + i) * 608 + cc] = d[i];
    }
  }
  __syncthreads();

  // ---- phase 2: integer shift-softmax (16 lanes per row) ----
  {
    int row = threadIdx.x >> 4, sub = threadIdx.x & 15;
    float sq = *s_qkv_slot;
    float sf = sq * sq * 0.125f;                             // * D^-0.5 (D=64)
    float x0f = floorf(-1.f / sf);
    int mx = -2147483647;
    for (int c = sub; c < Nn; c += 16) mx = max(mx, L[row * 608 + c]);
    for (int s = 8; s; s >>= 1) mx = max(mx, __shfl_xor(mx, s, 16));
    float* Lf = (float*)smemA;
    float sum = 0.f;
    for (int c = sub; c < Nn; c += 16) {
      int xi = L[row * 608 + c] - mx;
      xi = xi + (xi >> 1) - (xi >> 4);                       // x + floor(x/2) - floor(x/16)
      float xf = fmaxf((float)xi, 15.f * x0f);
      float qq = floorf(xf / x0f);
      float r = xf - x0f * qq;
      float ev = fmaxf(floorf((r * 0.5f - x0f) * exp2f(15.f - qq)), 0.f);
      Lf[row * 608 + c] = ev;
      sum += ev;
    }
    for (int s = 8; s; s >>= 1) sum += __shfl_xor(sum, s, 16);
    float esum = fminf(sum, 2147483647.f);
    float factor = floorf(2147483647.f / esum);
    for (int c = sub; c < Nn; c += 16) {
      float p = floorf(Lf[row * 608 + c] * factor * (1.f / 16777216.f)); // /2^(31-8+1)
      sProb[row * 640 + c] = (signed char)p;
    }
    for (int c = Nn + sub; c < 640; c += 16) sProb[row * 640 + c] = 0;
  }
  __syncthreads();

  // ---- phase 3: stage V transposed into LDS: VT[d][m] ----
  signed char* VT = (signed char*)smemA;                     // [64][640]
  for (int idx = threadIdx.x; idx < Nn * 16; idx += 256) {
    int m = idx >> 4, dg = (idx & 15) * 4;
    int wv = *(const int*)(vbase + (long)m * OC3 + dg);
    VT[(dg + 0) * 640 + m] = (signed char)(wv & 0xff);
    VT[(dg + 1) * 640 + m] = (signed char)((wv >> 8) & 0xff);
    VT[(dg + 2) * 640 + m] = (signed char)((wv >> 16) & 0xff);
    VT[(dg + 3) * 640 + m] = (signed char)((wv >> 24) & 0xff);
  }
  for (int idx = threadIdx.x; idx < 64 * (640 - Nn); idx += 256) {
    int d = idx / (640 - Nn), m = Nn + idx % (640 - Nn);
    VT[d * 640 + m] = 0;                                     // zero K padding
  }
  __syncthreads();

  // ---- phase 4: ctx = probs . v  (waves 0..3 each own one 16-wide D tile) ----
  unsigned am = 0;
  if (w < 4) {
    v8i acc = {};
    for (int ks = 0; ks < 10; ++ks) {                        // K = 640 (padded)
      v8i ap = frag_load_lds(sProb, 640, 0, ks * 64);
      v8i bv = frag_load_lds(VT, 640, w * 16, ks * 64);
      acc = WMMA_IU8(ap, bv, acc);
    }
    int rb = (lane >> 4) * 8;
    int cc = h * Dd + w * 16 + (lane & 15);
#pragma unroll
    for (int i = 0; i < 8; ++i) {
      int r = rt * 16 + rb + i;
      if (r < Nn) {
        int v = acc[i];
        ctx_i[((long)b * Nn + r) * Cc + cc] = v;
        unsigned a = (unsigned)(v < 0 ? -v : v);
        am = am > a ? am : a;
      }
    }
  }
  for (int s = 16; s; s >>= 1) { unsigned o = __shfl_xor(am, s, 32); am = am > o ? am : o; }
  if (lane == 0 && am) atomicMax(maxslot, am);
}

// ===================== launcher =====================
extern "C" void kernel_launch(void* const* d_in, const int* in_sizes, int n_in,
                              void* d_out, int out_size, void* d_ws, size_t ws_size,
                              hipStream_t stream) {
  const float* x      = (const float*)d_in[0];
  const float* w_qkv  = (const float*)d_in[1];
  const float* b_qkv  = (const float*)d_in[2];
  const float* w_proj = (const float*)d_in[3];
  const float* b_proj = (const float*)d_in[4];
  float* out = (float*)d_out;

  char* ws = (char*)d_ws;
  unsigned* scal = (unsigned*)(ws + OFF_SCAL);
  signed char* x8   = (signed char*)(ws + OFF_X8);
  signed char* wq8  = (signed char*)(ws + OFF_WQ8);
  signed char* wp8  = (signed char*)(ws + OFF_WP8);
  float* s_wq = (float*)(ws + OFF_SWQ);
  float* s_wp = (float*)(ws + OFF_SWP);
  int* qkv_i  = (int*)(ws + OFF_QKVI);
  signed char* qkv8 = (signed char*)(ws + OFF_QKV8);
  int* ctx_i  = (int*)(ws + OFF_CTXI);
  signed char* ctx8 = (signed char*)(ws + OFF_CTX8);

  const float* s_x_p   = (const float*)(scal + 1);
  const float* s_qkv_p = (const float*)(scal + 3);
  const float* s_ctx_p = (const float*)(scal + 5);
  long nX = (long)M_ROWS * Cc;

  k_init<<<1, 32, 0, stream>>>(scal);
  k_absmax<<<2048, 256, 0, stream>>>(x, nX, scal + 0);
  k_make_scale<<<1, 1, 0, stream>>>(scal + 0, 0, nullptr, 1.f, (float*)(scal + 1));
  k_wquant<<<(OC3 + 7) / 8, 256, 0, stream>>>(w_qkv, wq8, s_wq, OC3, Cc);
  k_wquant<<<(Cc + 7) / 8, 256, 0, stream>>>(w_proj, wp8, s_wp, Cc, Cc);
  k_quant_act<<<2048, 256, 0, stream>>>(x, x8, s_x_p, nX);

  // qkv = x8 @ wq8^T (int32 out + fused abs-max of dequantized value)
  k_gemm_i8<<<dim3((M_ROWS + 63) / 64, OC3 / 128), 256, 0, stream>>>(
      x8, wq8, M_ROWS, OC3, Cc, b_qkv, s_wq, s_x_p, qkv_i, nullptr, scal + 2, 0);
  k_make_scale<<<1, 1, 0, stream>>>(scal + 2, 0, nullptr, 1.f, (float*)(scal + 3));
  k_requant<<<2048, 256, 0, stream>>>(qkv_i, qkv8, s_x_p, s_wq, 1.f, s_qkv_p,
                                      (long)M_ROWS * OC3, OC3);

  // fused integer attention: logits -> shift-softmax -> probs @ v
  k_attn<<<dim3(37, Hh, Bb), 256, 0, stream>>>(qkv8, ctx_i, s_qkv_p, scal + 4);
  // s_ctx = max|ctx_int| * s_qkv * (1/128) / 127
  k_make_scale<<<1, 1, 0, stream>>>(scal + 4, 1, s_qkv_p, 0.0078125f, (float*)(scal + 5));
  k_requant<<<2048, 256, 0, stream>>>(ctx_i, ctx8, s_qkv_p, nullptr, 0.0078125f, s_ctx_p,
                                      nX, Cc);

  // out = dequant(ctx8 @ wp8^T + bias)
  k_gemm_i8<<<dim3((M_ROWS + 63) / 64, Cc / 128), 256, 0, stream>>>(
      ctx8, wp8, M_ROWS, Cc, Cc, b_proj, s_wp, s_ctx_p, nullptr, out, nullptr, 1);

  (void)in_sizes; (void)n_in; (void)out_size; (void)ws_size;
}
